// MinimalTransformerLayer_13176959664584
// MI455X (gfx1250) — matrix operand order repaired and checked by reading, not compiled, not run on hardware
//
#include <hip/hip_runtime.h>

// ---------------------------------------------------------------------------
// MI455X (gfx1250, wave32) transformer layer, bf16 WMMA everywhere:
//   qkv = x@Wqkv; flash attention over concat(past_kv, new_kv);
//   out-proj + residual; FFN (relu) + residual.
// Weights pre-transposed to [N,K] bf16 so A- and B-tiles are K-contiguous ->
// staged into LDS via GLOBAL_LOAD_ASYNC_TO_LDS_B128 (double buffered, waited
// with s_wait_asynccnt).
// ---------------------------------------------------------------------------

typedef __attribute__((ext_vector_type(16))) __bf16 v16bf;
typedef __attribute__((ext_vector_type(8)))  float  v8f;
typedef __attribute__((ext_vector_type(4)))  int    v4i;

#define GLOBAL_AS __attribute__((address_space(1)))
#define LDS_AS    __attribute__((address_space(3)))

#if defined(__gfx1250__) && __has_builtin(__builtin_amdgcn_global_load_async_to_lds_b128)
#define HAS_ASYNC_LDS 1
// copy 16 bytes global -> LDS, tracked by ASYNCcnt
#define ASYNC_CP16(g, l)                                             \
  __builtin_amdgcn_global_load_async_to_lds_b128(                    \
      (GLOBAL_AS v4i*)(g), (LDS_AS v4i*)(l), 0, 0)
#if __has_builtin(__builtin_amdgcn_s_wait_asynccnt)
#define WAIT_ASYNC(n) __builtin_amdgcn_s_wait_asynccnt(n)
#else
#define WAIT_ASYNC(n) asm volatile("s_wait_asynccnt %0" ::"i"(n) : "memory")
#endif
#else
#define HAS_ASYNC_LDS 0
#define ASYNC_CP16(g, l) (*(uint4*)(l) = *(const uint4*)(g))
#define WAIT_ASYNC(n)
#endif

union Frag {
  v16bf v;
  unsigned int u[8];
};

// fp32 -> bf16 (round to nearest even), stored as ushort
__device__ inline unsigned short f2b(float f) {
  unsigned int u = __float_as_uint(f);
  unsigned int r = u + 0x7FFFu + ((u >> 16) & 1u);
  return (unsigned short)(r >> 16);
}

// -------- WMMA fragment loaders (per CDNA5 ISA 7.12.2 layouts) --------------
// A-matrix 16x32 bf16: lanes 0-15 -> rows 0-15 with K={0..7,16..23},
//                      lanes 16-31 -> rows 0-15 with K={8..15,24..31}
__device__ inline v16bf load_frag_a(const unsigned short* smem, int ld) {
  int lane = threadIdx.x & 31;
  int row = lane & 15;
  int kb = (lane < 16) ? 0 : 8;
  Frag f;
  const unsigned short* p = smem + row * ld;
#pragma unroll
  for (int v = 0; v < 4; ++v)
    f.u[v] = *(const unsigned int*)(p + kb + 2 * v);
#pragma unroll
  for (int v = 0; v < 4; ++v)
    f.u[4 + v] = *(const unsigned int*)(p + kb + 16 + 2 * v);
  return f.v;
}

// B-matrix 32x16 bf16, staged K-contiguous per column: smem[n*ld + k].
// lanes 0-15 -> cols 0-15 with K={0..7,16..23}; lanes 16-31 -> K={8..15,24..31}
__device__ inline v16bf load_frag_b(const unsigned short* smem, int ld) {
  int lane = threadIdx.x & 31;
  int n = lane & 15;
  int kb = (lane < 16) ? 0 : 8;
  Frag f;
  const unsigned short* p = smem + n * ld + kb;
#pragma unroll
  for (int v = 0; v < 4; ++v)
    f.u[v] = *(const unsigned int*)(p + 2 * v);
#pragma unroll
  for (int v = 0; v < 4; ++v)
    f.u[4 + v] = *(const unsigned int*)(p + 16 + 2 * v);
  return f.v;
}

// ------------------------------ elementwise ---------------------------------
__global__ void f32_to_bf16_kernel(const float* __restrict__ in,
                                   unsigned short* __restrict__ out, int n) {
  int i = blockIdx.x * blockDim.x + threadIdx.x;
  if (i < n) out[i] = f2b(in[i]);
}

// W [Kd,Nd] fp32 -> Wt [Nd,Kd] bf16, tiled 32x32 through LDS (both sides
// coalesced). Makes every GEMM tile K-contiguous for async LDS staging.
__global__ __launch_bounds__(256) void transpose_to_bf16_kernel(
    const float* __restrict__ W, unsigned short* __restrict__ Wt, int Kd,
    int Nd) {
  __shared__ float tile[32][33];
  int n0 = blockIdx.x * 32;
  int k0 = blockIdx.y * 32;
  int tx = threadIdx.x & 31;
  int ty = threadIdx.x >> 5;  // 0..7
#pragma unroll
  for (int j = 0; j < 32; j += 8)
    tile[ty + j][tx] = W[(size_t)(k0 + ty + j) * Nd + n0 + tx];
  __syncthreads();
#pragma unroll
  for (int j = 0; j < 32; j += 8)
    Wt[(size_t)(n0 + ty + j) * Kd + k0 + tx] = f2b(tile[tx][ty + j]);
}

// Concatenate past_k/past_v with new k/v (from qkv fp32), writing fp32 outputs
// and bf16 copies for the attention kernel in one pass.
__global__ void concat_kv_kernel(const float* __restrict__ past_k,
                                 const float* __restrict__ past_v,
                                 const float* __restrict__ qkv,
                                 float* __restrict__ k_out,
                                 float* __restrict__ v_out,
                                 unsigned short* __restrict__ kb,
                                 unsigned short* __restrict__ vb) {
  const int n = 4 * 16 * 2560 * 128;
  int i = blockIdx.x * blockDim.x + threadIdx.x;
  if (i >= n) return;
  int d = i & 127;
  int p = (i >> 7) % 2560;
  int h = (i / (128 * 2560)) & 15;
  int b = i / (128 * 2560 * 16);
  float kvv, vvv;
  if (p < 2048) {
    int src = ((b * 16 + h) * 2048 + p) * 128 + d;
    kvv = past_k[src];
    vvv = past_v[src];
  } else {
    size_t row = (size_t)b * 512 + (p - 2048);
    kvv = qkv[row * 6144 + 2048 + h * 128 + d];
    vvv = qkv[row * 6144 + 4096 + h * 128 + d];
  }
  k_out[i] = kvv;
  v_out[i] = vvv;
  kb[i] = f2b(kvv);
  vb[i] = f2b(vvv);
}

// q slice of qkv -> head-major bf16 [B,NH,S,HD]
__global__ void q_to_heads_kernel(const float* __restrict__ qkv,
                                  unsigned short* __restrict__ qb) {
  const int n = 4 * 16 * 512 * 128;
  int i = blockIdx.x * blockDim.x + threadIdx.x;
  if (i >= n) return;
  int d = i & 127;
  int s = (i >> 7) & 511;
  int h = (i >> 16) & 15;
  int b = i >> 20;
  qb[i] = f2b(qkv[((size_t)b * 512 + s) * 6144 + h * 128 + d]);
}

// ------------------------------- GEMM ---------------------------------------
// C[M,N] = A[M,K](bf16) @ Wt[N,K](bf16, pre-transposed)
//          (+res) (relu) -> outf fp32 / outb bf16
// 128x128 block tile, K-step 32, double-buffered async LDS staging,
// 8 waves each computing a 32x64 sub-tile (2x4 grid of WMMA accumulators).
__global__ __launch_bounds__(256) void gemm_bf16_kernel(
    const unsigned short* __restrict__ A, const unsigned short* __restrict__ Wt,
    const float* __restrict__ res, float* __restrict__ outf,
    unsigned short* __restrict__ outb, int M, int N, int K, int relu) {
  __shared__ __align__(16) unsigned short Alds[2][128 * 32];  // [row][k]
  __shared__ __align__(16) unsigned short Wlds[2][128 * 32];  // [n][k]

  int tid = threadIdx.x;
  int lane = tid & 31;
  int wave = tid >> 5;
  int wm = wave >> 1;  // 4 row groups of 32 rows
  int wn = wave & 1;   // 2 col groups of 64 cols
  int rb = blockIdx.y * 128;
  int cb = blockIdx.x * 128;
  int r = tid >> 1;    // staged row (of A tile and of Wt tile)
  int half = tid & 1;  // 32B half of the 64B row

  v8f acc[2][4];
#pragma unroll
  for (int mi = 0; mi < 2; ++mi)
#pragma unroll
    for (int ni = 0; ni < 4; ++ni) acc[mi][ni] = {};

  // issue one 32-deep K tile (A + W) into LDS buffer `buf`: 4 async B128/thread
  auto stage = [&](int kt, int buf) {
    const unsigned short* ga = A + (size_t)(rb + r) * K + kt * 32 + half * 16;
    const unsigned short* gw = Wt + (size_t)(cb + r) * K + kt * 32 + half * 16;
    unsigned short* la = &Alds[buf][r * 32 + half * 16];
    unsigned short* lw = &Wlds[buf][r * 32 + half * 16];
    ASYNC_CP16(ga, la);
    ASYNC_CP16(ga + 8, la + 8);
    ASYNC_CP16(gw, lw);
    ASYNC_CP16(gw + 8, lw + 8);
  };

  int nk = K >> 5;
  stage(0, 0);
  for (int i = 0; i < nk; ++i) {
    int cur = i & 1;
    if (i + 1 < nk) {
      stage(i + 1, cur ^ 1);
      WAIT_ASYNC(4);  // tile i's 4 async ops retired (in-order completion)
    } else {
      WAIT_ASYNC(0);
    }
    __syncthreads();

    v16bf af[2], bfr[4];
#pragma unroll
    for (int mi = 0; mi < 2; ++mi)
      af[mi] = load_frag_a(&Alds[cur][(wm * 32 + mi * 16) * 32], 32);
#pragma unroll
    for (int ni = 0; ni < 4; ++ni)
      bfr[ni] = load_frag_b(&Wlds[cur][(wn * 64 + ni * 16) * 32], 32);
#pragma unroll
    for (int mi = 0; mi < 2; ++mi)
#pragma unroll
      for (int ni = 0; ni < 4; ++ni)
        acc[mi][ni] = __builtin_amdgcn_wmma_f32_16x16x32_bf16(
            false, af[mi], false, bfr[ni], (short)0, acc[mi][ni], false, false);
    __syncthreads();  // tile fully consumed before its buffer is re-staged
  }

  // epilogue: C layout -> vgpr j holds row j (lanes 0-15) / row 8+j (16-31)
  int hi8 = (lane < 16) ? 0 : 8;
  int c16 = lane & 15;
#pragma unroll
  for (int mi = 0; mi < 2; ++mi)
#pragma unroll
    for (int ni = 0; ni < 4; ++ni)
#pragma unroll
      for (int j = 0; j < 8; ++j) {
        int row = rb + wm * 32 + mi * 16 + hi8 + j;
        int col = cb + wn * 64 + ni * 16 + c16;
        float v = acc[mi][ni][j];
        if (res) v += res[(size_t)row * N + col];
        if (relu) v = fmaxf(v, 0.0f);
        if (outf) outf[(size_t)row * N + col] = v;
        if (outb) outb[(size_t)row * N + col] = f2b(v);
      }
}

// ---------------------------- flash attention -------------------------------
// Block: one (b,h) and a 128-row q tile; 8 waves, each owns 16 q rows.
// Loop 20 kv-chunks of 128: scores = Q@K^T via WMMA, online softmax (row
// reductions across 16-lane halves), P re-laid out C->A via per-wave LDS,
// O += P@V via WMMA. Q and K chunks staged via async global->LDS.
__global__ __launch_bounds__(256) void attention_kernel(
    const unsigned short* __restrict__ Q, const unsigned short* __restrict__ Kc,
    const unsigned short* __restrict__ Vc, unsigned short* __restrict__ O) {
  __shared__ __align__(16) unsigned short Qs[128 * 128];    // [qrow][d]
  __shared__ __align__(16) unsigned short Ks[128 * 128];    // [kvrow][d]
  __shared__ __align__(16) unsigned short Vts[128 * 128];   // [d][kvrow]
  __shared__ __align__(16) unsigned short Ps[8][16 * 128];  // per-wave P tile

  int tid = threadIdx.x;
  int lane = tid & 31;
  int wv = tid >> 5;
  int bid = blockIdx.x;
  int qt = bid & 3;
  int h = (bid >> 2) & 15;
  int b = bid >> 6;
  int q0 = qt * 128;
  int hi8 = (lane < 16) ? 0 : 8;
  int c16 = lane & 15;
  const float rscale = 0.088388347648318447f;  // 1/sqrt(128)

  // stage Q tile (32KB, contiguous): 128B per thread, async
  const size_t qbase = (((size_t)b * 16 + h) * 512 + q0) * 128;
  {
    const unsigned short* src = Q + qbase + tid * 64;
    unsigned short* dst = Qs + tid * 64;
#pragma unroll
    for (int j = 0; j < 8; ++j) ASYNC_CP16(src + j * 8, dst + j * 8);
  }
  WAIT_ASYNC(0);
  __syncthreads();

  v16bf qf[4];
#pragma unroll
  for (int kd = 0; kd < 4; ++kd)
    qf[kd] = load_frag_a(Qs + (wv * 16) * 128 + kd * 32, 128);

  v8f o[8];
#pragma unroll
  for (int t = 0; t < 8; ++t) o[t] = {};
  float mrun[8], lrun[8];
#pragma unroll
  for (int i = 0; i < 8; ++i) {
    mrun[i] = -3.0e38f;
    lrun[i] = 0.0f;
  }

  const size_t kvbase = (((size_t)b * 16 + h) * 2560) * 128;

  for (int kc = 0; kc < 20; ++kc) {
    __syncthreads();
    {  // stage K chunk [128][128] via async (overlaps the V transpose below)
      const unsigned short* src = Kc + kvbase + (size_t)kc * 128 * 128 + tid * 64;
      unsigned short* dst = Ks + tid * 64;
#pragma unroll
      for (int j = 0; j < 8; ++j) ASYNC_CP16(src + j * 8, dst + j * 8);
    }
    {  // stage V chunk transposed -> Vts[d][kv] (needs VALU repack)
      const unsigned short* vsrc = Vc + kvbase + (size_t)kc * 128 * 128;
      for (int i = tid; i < 4096; i += 256) {
        int rr = i >> 5;        // kv row
        int d0 = (i & 31) * 4;  // d start
        uint2 u = *(const uint2*)(vsrc + rr * 128 + d0);
        Vts[(d0 + 0) * 128 + rr] = (unsigned short)(u.x & 0xffff);
        Vts[(d0 + 1) * 128 + rr] = (unsigned short)(u.x >> 16);
        Vts[(d0 + 2) * 128 + rr] = (unsigned short)(u.y & 0xffff);
        Vts[(d0 + 3) * 128 + rr] = (unsigned short)(u.y >> 16);
      }
    }
    WAIT_ASYNC(0);
    __syncthreads();

    // scores: 8 column tiles of 16 kv
    v8f sc[8];
#pragma unroll
    for (int t = 0; t < 8; ++t) sc[t] = {};
#pragma unroll
    for (int kd = 0; kd < 4; ++kd)
#pragma unroll
      for (int t = 0; t < 8; ++t) {
        v16bf bk = load_frag_b(Ks + (t * 16) * 128 + kd * 32, 128);
        sc[t] = __builtin_amdgcn_wmma_f32_16x16x32_bf16(
            false, qf[kd], false, bk, (short)0, sc[t], false, false);
      }

    // online softmax; row i lives in vgpr i across the 16 lanes of each half
    unsigned short* pw = &Ps[wv][0];
#pragma unroll
    for (int i = 0; i < 8; ++i) {
      float pv[8];
      float rmax = -3.0e38f;
#pragma unroll
      for (int t = 0; t < 8; ++t) {
        pv[t] = sc[t][i] * rscale;
        rmax = fmaxf(rmax, pv[t]);
      }
#pragma unroll
      for (int off = 1; off < 16; off <<= 1)
        rmax = fmaxf(rmax, __shfl_xor(rmax, off, 32));
      float mnew = fmaxf(mrun[i], rmax);
      float alpha = __expf(mrun[i] - mnew);
      float rsum = 0.0f;
#pragma unroll
      for (int t = 0; t < 8; ++t) {
        float p = __expf(pv[t] - mnew);
        rsum += p;
        pw[(i + hi8) * 128 + t * 16 + c16] = f2b(p);
      }
#pragma unroll
      for (int off = 1; off < 16; off <<= 1) rsum += __shfl_xor(rsum, off, 32);
      lrun[i] = lrun[i] * alpha + rsum;
      mrun[i] = mnew;
#pragma unroll
      for (int t = 0; t < 8; ++t) o[t][i] = o[t][i] * alpha;
    }

    // O += P @ V  (A frags re-read from per-wave LDS; same-wave LDS in-order)
#pragma unroll
    for (int kvd = 0; kvd < 4; ++kvd) {
      v16bf pa = load_frag_a(pw + kvd * 32, 128);
#pragma unroll
      for (int t = 0; t < 8; ++t) {
        v16bf bv = load_frag_b(Vts + (t * 16) * 128 + kvd * 32, 128);
        o[t] = __builtin_amdgcn_wmma_f32_16x16x32_bf16(
            false, pa, false, bv, (short)0, o[t], false, false);
      }
    }
  }

  // normalize + store merged-head bf16 [B*S, H]
#pragma unroll
  for (int t = 0; t < 8; ++t)
#pragma unroll
    for (int j = 0; j < 8; ++j) {
      int row = q0 + wv * 16 + hi8 + j;
      float val = o[t][j] / lrun[j];
      O[((size_t)b * 512 + row) * 2048 + h * 128 + t * 16 + c16] = f2b(val);
    }
}

// ------------------------------- launcher -----------------------------------
extern "C" void kernel_launch(void* const* d_in, const int* in_sizes, int n_in,
                              void* d_out, int out_size, void* d_ws,
                              size_t ws_size, hipStream_t stream) {
  const float* x      = (const float*)d_in[0];
  const float* past_k = (const float*)d_in[1];
  const float* past_v = (const float*)d_in[2];
  const float* Wqkv   = (const float*)d_in[3];
  const float* Wout   = (const float*)d_in[4];
  const float* W1     = (const float*)d_in[5];
  const float* W2     = (const float*)d_in[6];

  float* xo = (float*)d_out;                // [4,512,2048]
  float* ko = xo + (size_t)4 * 512 * 2048;  // [4,16,2560,128]
  float* vo = ko + (size_t)4 * 16 * 2560 * 128;

  char* wp = (char*)d_ws;
  auto alloc = [&](size_t bytes) {
    char* p = wp;
    wp += (bytes + 255) & ~(size_t)255;
    return p;
  };
  unsigned short* xb    = (unsigned short*)alloc((size_t)2048 * 2048 * 2);
  unsigned short* wqkvb = (unsigned short*)alloc((size_t)2048 * 6144 * 2);  // [N,K]
  unsigned short* woutb = (unsigned short*)alloc((size_t)2048 * 2048 * 2);  // [N,K]
  unsigned short* w1b   = (unsigned short*)alloc((size_t)2048 * 8192 * 2);  // [N,K]
  unsigned short* w2b   = (unsigned short*)alloc((size_t)8192 * 2048 * 2);  // [N,K]
  float*          qkvf  = (float*)alloc((size_t)2048 * 6144 * 4);
  unsigned short* qb    = (unsigned short*)alloc((size_t)4 * 16 * 512 * 128 * 2);
  unsigned short* kbm   = (unsigned short*)alloc((size_t)4 * 16 * 2560 * 128 * 2);
  unsigned short* vbm   = (unsigned short*)alloc((size_t)4 * 16 * 2560 * 128 * 2);
  unsigned short* ob    = (unsigned short*)alloc((size_t)2048 * 2048 * 2);
  unsigned short* x1b   = (unsigned short*)alloc((size_t)2048 * 2048 * 2);
  unsigned short* h1b   = (unsigned short*)alloc((size_t)2048 * 8192 * 2);

  // activations: plain convert (consumed row-major as GEMM A)
  f32_to_bf16_kernel<<<(2048 * 2048 + 255) / 256, 256, 0, stream>>>(
      x, xb, 2048 * 2048);
  // weights: transpose-convert to [N,K] so GEMM tiles are K-contiguous
  transpose_to_bf16_kernel<<<dim3(6144 / 32, 2048 / 32), 256, 0, stream>>>(
      Wqkv, wqkvb, 2048, 6144);
  transpose_to_bf16_kernel<<<dim3(2048 / 32, 2048 / 32), 256, 0, stream>>>(
      Wout, woutb, 2048, 2048);
  transpose_to_bf16_kernel<<<dim3(8192 / 32, 2048 / 32), 256, 0, stream>>>(
      W1, w1b, 2048, 8192);
  transpose_to_bf16_kernel<<<dim3(2048 / 32, 8192 / 32), 256, 0, stream>>>(
      W2, w2b, 8192, 2048);

  // qkv = x @ Wqkv  (fp32 out for kv concat / q extraction)
  gemm_bf16_kernel<<<dim3(6144 / 128, 2048 / 128), 256, 0, stream>>>(
      xb, wqkvb, nullptr, qkvf, nullptr, 2048, 6144, 2048, 0);

  // concat past kv + new kv -> fp32 outputs + bf16 copies
  concat_kv_kernel<<<(4 * 16 * 2560 * 128 + 255) / 256, 256, 0, stream>>>(
      past_k, past_v, qkvf, ko, vo, kbm, vbm);

  q_to_heads_kernel<<<(4 * 16 * 512 * 128 + 255) / 256, 256, 0, stream>>>(qkvf,
                                                                          qb);

  attention_kernel<<<256, 256, 0, stream>>>(qb, kbm, vbm, ob);

  // x1 = x + attn_out @ Wout  -> fp32 into d_out, bf16 copy for FFN
  gemm_bf16_kernel<<<dim3(2048 / 128, 2048 / 128), 256, 0, stream>>>(
      ob, woutb, x, xo, x1b, 2048, 2048, 2048, 0);

  // h1 = relu(x1 @ W1)  (bf16 only)
  gemm_bf16_kernel<<<dim3(8192 / 128, 2048 / 128), 256, 0, stream>>>(
      x1b, w1b, nullptr, nullptr, h1b, 2048, 8192, 2048, 1);

  // x2 = x1 + h1 @ W2  -> fp32 into d_out (read+overwrite of same region)
  gemm_bf16_kernel<<<dim3(2048 / 128, 2048 / 128), 256, 0, stream>>>(
      h1b, w2b, xo, xo, nullptr, 2048, 2048, 8192, 0);
}
